// TransformerBlock_9028021256566
// MI455X (gfx1250) — compile-verified, matrix operand order checked by reading, not run on hardware
//
#include <hip/hip_runtime.h>
#include <hip/hip_bf16.h>
#include <math.h>

typedef _Float16 half_t;
typedef __attribute__((ext_vector_type(4)))  _Float16 v4h;
typedef __attribute__((ext_vector_type(8)))  _Float16 v8h;
typedef __attribute__((ext_vector_type(16))) _Float16 v16h;
typedef __attribute__((ext_vector_type(4)))  float    v4f;
typedef __attribute__((ext_vector_type(8)))  float    v8f;

#define DIM      384
#define HEADS    6
#define HEAD_DIM 64
#define MLP_DIM  1536
#define SEQ      2048
#define BATCH    8
#define TOKENS   (BATCH * SEQ)   // 16384
#define QKV_LD   (3 * DIM)       // 1152

// ---------------------------------------------------------------------------
// WMMA helpers
// ---------------------------------------------------------------------------
__device__ __forceinline__ v8f wmma_f16(v16h a, v16h b, v8f c) {
  return __builtin_amdgcn_wmma_f32_16x16x32_f16(
      /*neg_a=*/false, a, /*neg_b=*/false, b,
      /*c_mod=*/(short)0, c, /*reuse_a=*/false, /*reuse_b=*/false);
}

union tile_u { v16h v; v8h h[2]; };

// Load one 16x32 f16 operand tile (A or BT layout) from a row-major matrix.
// `base` stays a uniform kernel-arg pointer (SGPR); `off` is a 32-bit per-lane
// element offset -> global_load_b128 saddr+voffset, loading directly into the
// contiguous VGPR quad pair (no packing moves).
__device__ __forceinline__ v16h load_tile_off(const half_t* __restrict__ base,
                                              int off) {
  tile_u u;
  u.h[0] = *(const v8h*)(base + off);
  u.h[1] = *(const v8h*)(base + off + 16);
  return u.v;
}

__device__ __forceinline__ float grp16_max(float v) {
  #pragma unroll
  for (int m = 8; m >= 1; m >>= 1) v = fmaxf(v, __shfl_xor(v, m, 32));
  return v;
}
__device__ __forceinline__ float grp16_sum(float v) {
  #pragma unroll
  for (int m = 8; m >= 1; m >>= 1) v += __shfl_xor(v, m, 32);
  return v;
}
__device__ __forceinline__ float wave_sum(float v) {
  #pragma unroll
  for (int m = 16; m >= 1; m >>= 1) v += __shfl_xor(v, m, 32);
  return v;
}
__device__ __forceinline__ float gelu_exact(float x) {
  return 0.5f * x * (1.0f + erff(x * 0.70710678118654752440f));
}

// ---------------------------------------------------------------------------
// Weight convert + transpose: W[K,N] f32 -> WT[N,K] f16
// ---------------------------------------------------------------------------
__global__ void cvt_transpose_kernel(const float* __restrict__ w,
                                     half_t* __restrict__ wt, int K, int N) {
  int idx = blockIdx.x * blockDim.x + threadIdx.x;
  if (idx >= K * N) return;
  int n = idx / K;
  int k = idx - n * K;
  wt[idx] = (half_t)w[(size_t)k * N + n];
}

// ---------------------------------------------------------------------------
// LayerNorm over C=384, one wave per token, f16 output.
// Vectorized: 3x float4 loads, 3x 4xf16 stores per lane.
// ---------------------------------------------------------------------------
__global__ void __launch_bounds__(256) ln_f16_kernel(
    const float* __restrict__ x, const float* __restrict__ g,
    const float* __restrict__ b, half_t* __restrict__ out) {
  int wid = threadIdx.x >> 5, lane = threadIdx.x & 31;
  int row = blockIdx.x * 8 + wid;
  int rb = row * DIM;
  v4f v[3], gv[3], bv[3];
  float s = 0.f;
  #pragma unroll
  for (int i = 0; i < 3; ++i) {
    int c = (lane + 32 * i) * 4;
    v[i]  = *(const v4f*)(x + rb + c);
    gv[i] = *(const v4f*)(g + c);
    bv[i] = *(const v4f*)(b + c);
    s += v[i][0] + v[i][1] + v[i][2] + v[i][3];
  }
  s = wave_sum(s);
  float mu = s * (1.0f / DIM);
  float vs = 0.f;
  #pragma unroll
  for (int i = 0; i < 3; ++i) {
    #pragma unroll
    for (int k = 0; k < 4; ++k) { float d = v[i][k] - mu; vs += d * d; }
  }
  vs = wave_sum(vs);
  float rs = rsqrtf(vs * (1.0f / DIM) + 1e-5f);
  #pragma unroll
  for (int i = 0; i < 3; ++i) {
    int c = (lane + 32 * i) * 4;
    v4h o;
    #pragma unroll
    for (int k = 0; k < 4; ++k)
      o[k] = (half_t)((v[i][k] - mu) * rs * gv[i][k] + bv[i][k]);
    *(v4h*)(out + rb + c) = o;
  }
}

// ---------------------------------------------------------------------------
// V transpose: qkv_h [B*N,1152] (V at +768) -> vt [B*H][D=64][N=2048] f16
// ---------------------------------------------------------------------------
__global__ void transpose_v_kernel(const half_t* __restrict__ qkv,
                                   half_t* __restrict__ vt) {
  int idx = blockIdx.x * blockDim.x + threadIdx.x;  // < 48*64*2048
  int n  = idx & (SEQ - 1);
  int d  = (idx >> 11) & 63;
  int bh = idx >> 17;
  int b  = bh / HEADS, h = bh - b * HEADS;
  vt[idx] = qkv[(b * SEQ + n) * QKV_LD + 2 * DIM + h * HEAD_DIM + d];
}

// ---------------------------------------------------------------------------
// WMMA GEMM: C[M,N] = act(A[M,K]f16 * BT[N,K]f16^T + bias) (+resid)
// 4 waves/block, each wave 64x64 (16 accumulator tiles), K step 32.
// ---------------------------------------------------------------------------
template <int N, int K, bool OUTF, bool RESID, bool GELU>
__global__ void __launch_bounds__(128) gemm_wmma_kernel(
    const half_t* __restrict__ A, const half_t* __restrict__ BT,
    const float* __restrict__ bias, const float* __restrict__ resid,
    float* __restrict__ outF, half_t* __restrict__ outH) {
  int lane = threadIdx.x & 31;
  int wid  = threadIdx.x >> 5;
  int r    = lane & 15;
  int sel  = lane >> 4;
  int row0 = blockIdx.y * 128 + (wid >> 1) * 64;
  int col0 = blockIdx.x * 128 + (wid & 1) * 64;

  int aoff[4], boff[4];
  #pragma unroll
  for (int i = 0; i < 4; ++i) aoff[i] = (row0 + i * 16 + r) * K + sel * 8;
  #pragma unroll
  for (int i = 0; i < 4; ++i) boff[i] = (col0 + i * 16 + r) * K + sel * 8;

  v8f acc[4][4] = {};
  for (int k0 = 0; k0 < K; k0 += 32) {
    if (k0 + 64 < K) {
      __builtin_prefetch(A + aoff[0] + k0 + 64, 0, 0);
      __builtin_prefetch(BT + boff[0] + k0 + 64, 0, 0);
    }
    v16h a[4], bm[4];
    #pragma unroll
    for (int i = 0; i < 4; ++i) a[i]  = load_tile_off(A,  aoff[i] + k0);
    #pragma unroll
    for (int i = 0; i < 4; ++i) bm[i] = load_tile_off(BT, boff[i] + k0);
    #pragma unroll
    for (int mi = 0; mi < 4; ++mi)
      #pragma unroll
      for (int ni = 0; ni < 4; ++ni)
        acc[mi][ni] = wmma_f16(a[mi], bm[ni], acc[mi][ni]);
  }

  int n  = lane & 15;
  int mb = sel * 8;
  #pragma unroll
  for (int ni = 0; ni < 4; ++ni) {
    int col = col0 + ni * 16 + n;
    float bv = bias[col];
    #pragma unroll
    for (int mi = 0; mi < 4; ++mi) {
      int rb = (row0 + mi * 16 + mb) * N + col;
      #pragma unroll
      for (int rr = 0; rr < 8; ++rr) {
        float v = acc[mi][ni][rr] + bv;
        if (GELU) v = gelu_exact(v);
        int o = rb + rr * N;
        if (OUTF) {
          float res = RESID ? resid[o] : 0.0f;
          outF[o] = v + res;
        } else {
          outH[o] = (half_t)v;
        }
      }
    }
  }
}

// ---------------------------------------------------------------------------
// Flash attention: grid (N/64, B*H), 4 waves/block, each wave 16 query rows.
// S = (Q K^T)/8 via WMMA, online softmax, O += P V via WMMA (P staged in LDS
// for C-layout -> A-layout conversion; V^T pre-staged in global as f16).
// ---------------------------------------------------------------------------
__global__ void __launch_bounds__(128) flash_attn_kernel(
    const half_t* __restrict__ qkv, const half_t* __restrict__ vt,
    half_t* __restrict__ out /* [B*N, 384] f16 */) {
  __shared__ __align__(16) half_t pbuf[4][16][40];  // per-wave 16x32 P tile
  int lane = threadIdx.x & 31;
  int wid  = threadIdx.x >> 5;
  int r    = lane & 15;
  int sel  = lane >> 4;
  int bh   = blockIdx.y;
  int b    = bh / HEADS, h = bh - b * HEADS;
  int q0   = blockIdx.x * 64 + wid * 16;

  int laneK = r * QKV_LD + sel * 8;               // lane part for q/k tiles
  int qoff  = (b * SEQ + q0) * QKV_LD + h * HEAD_DIM + laneK;
  int kbase = b * SEQ * QKV_LD + DIM + h * HEAD_DIM + laneK;
  int vbase = bh * HEAD_DIM * SEQ + r * SEQ + sel * 8;

  v16h qa0 = load_tile_off(qkv, qoff);        // K dim 0..31
  v16h qa1 = load_tile_off(qkv, qoff + 32);   // K dim 32..63

  v8f o0 = {}, o1 = {}, o2 = {}, o3 = {};
  float m_[8], l_[8];
  #pragma unroll
  for (int i = 0; i < 8; ++i) { m_[i] = -1e30f; l_[i] = 0.0f; }

  int n  = lane & 15;
  int mb = sel * 8;
  half_t* prow = &pbuf[wid][0][0];

  for (int j = 0; j < SEQ; j += 32) {
    int kj = kbase + j * QKV_LD;
    if (j + 32 < SEQ) {
      __builtin_prefetch(qkv + kj + 32 * QKV_LD, 0, 0);
      __builtin_prefetch(vt + vbase + j + 32, 0, 0);
    }
    v16h k00 = load_tile_off(qkv, kj);
    v16h k01 = load_tile_off(qkv, kj + 32);
    v16h k10 = load_tile_off(qkv, kj + 16 * QKV_LD);
    v16h k11 = load_tile_off(qkv, kj + 16 * QKV_LD + 32);

    v8f s0 = {}; s0 = wmma_f16(qa0, k00, s0); s0 = wmma_f16(qa1, k01, s0);
    v8f s1 = {}; s1 = wmma_f16(qa0, k10, s1); s1 = wmma_f16(qa1, k11, s1);

    float corr[8];
    #pragma unroll
    for (int rr = 0; rr < 8; ++rr) {
      float a0 = s0[rr] * 0.125f, a1 = s1[rr] * 0.125f;
      float mx = grp16_max(fmaxf(a0, a1));
      float nm = fmaxf(m_[rr], mx);
      float sc = __expf(m_[rr] - nm);
      float p0 = __expf(a0 - nm), p1 = __expf(a1 - nm);
      float rs = grp16_sum(p0 + p1);
      l_[rr] = l_[rr] * sc + rs;
      m_[rr] = nm;
      corr[rr] = sc;
      prow[(mb + rr) * 40 + n]      = (half_t)p0;
      prow[(mb + rr) * 40 + n + 16] = (half_t)p1;
    }
    #pragma unroll
    for (int rr = 0; rr < 8; ++rr) {
      o0[rr] *= corr[rr]; o1[rr] *= corr[rr];
      o2[rr] *= corr[rr]; o3[rr] *= corr[rr];
    }

    // Re-read P from LDS in A-matrix layout (16x32): ds_load_b128 pairs
    {
      const half_t* pp = prow + r * 40 + sel * 8;
      tile_u u;
      u.h[0] = *(const v8h*)pp;
      u.h[1] = *(const v8h*)(pp + 16);

      int vj = vbase + j;
      v16h vb0 = load_tile_off(vt, vj);
      v16h vb1 = load_tile_off(vt, vj + 16 * SEQ);
      v16h vb2 = load_tile_off(vt, vj + 32 * SEQ);
      v16h vb3 = load_tile_off(vt, vj + 48 * SEQ);
      o0 = wmma_f16(u.v, vb0, o0);
      o1 = wmma_f16(u.v, vb1, o1);
      o2 = wmma_f16(u.v, vb2, o2);
      o3 = wmma_f16(u.v, vb3, o3);
    }
  }

  float inv[8];
  #pragma unroll
  for (int rr = 0; rr < 8; ++rr) inv[rr] = 1.0f / l_[rr];
  int ob = (b * SEQ + q0) * DIM + h * HEAD_DIM + n;
  #pragma unroll
  for (int rr = 0; rr < 8; ++rr) {
    int ro = ob + (mb + rr) * DIM;
    out[ro + 0]  = (half_t)(o0[rr] * inv[rr]);
    out[ro + 16] = (half_t)(o1[rr] * inv[rr]);
    out[ro + 32] = (half_t)(o2[rr] * inv[rr]);
    out[ro + 48] = (half_t)(o3[rr] * inv[rr]);
  }
}

// ---------------------------------------------------------------------------
extern "C" void kernel_launch(void* const* d_in, const int* in_sizes, int n_in,
                              void* d_out, int out_size, void* d_ws, size_t ws_size,
                              hipStream_t stream) {
  const float* x      = (const float*)d_in[0];
  const float* ln1_g  = (const float*)d_in[1];
  const float* ln1_b  = (const float*)d_in[2];
  const float* qkv_w  = (const float*)d_in[3];
  const float* qkv_b  = (const float*)d_in[4];
  const float* proj_w = (const float*)d_in[5];
  const float* proj_b = (const float*)d_in[6];
  const float* ln2_g  = (const float*)d_in[7];
  const float* ln2_b  = (const float*)d_in[8];
  const float* fc1_w  = (const float*)d_in[9];
  const float* fc1_b  = (const float*)d_in[10];
  const float* fc2_w  = (const float*)d_in[11];
  const float* fc2_b  = (const float*)d_in[12];
  float* out = (float*)d_out;

  size_t off = 0;
  auto take = [&](size_t bytes) -> void* {
    void* p = (char*)d_ws + off;
    off += (bytes + 255) & ~(size_t)255;
    return p;
  };
  half_t* qkvT  = (half_t*)take((size_t)QKV_LD * DIM * 2);
  half_t* projT = (half_t*)take((size_t)DIM * DIM * 2);
  half_t* fc1T  = (half_t*)take((size_t)MLP_DIM * DIM * 2);
  half_t* fc2T  = (half_t*)take((size_t)DIM * MLP_DIM * 2);
  half_t* hln   = (half_t*)take((size_t)TOKENS * DIM * 2);
  half_t* qkv_h = (half_t*)take((size_t)TOKENS * QKV_LD * 2);
  half_t* vtb   = (half_t*)take((size_t)BATCH * HEADS * HEAD_DIM * SEQ * 2);
  half_t* att_o = (half_t*)take((size_t)TOKENS * DIM * 2);
  float*  x1    = (float*) take((size_t)TOKENS * DIM * 4);
  half_t* hmlp  = (half_t*)take((size_t)TOKENS * MLP_DIM * 2);

  // --- weight f16 transposes (tiny) ---
  cvt_transpose_kernel<<<(DIM * QKV_LD + 255) / 256, 256, 0, stream>>>(qkv_w, qkvT, DIM, QKV_LD);
  cvt_transpose_kernel<<<(DIM * DIM + 255) / 256, 256, 0, stream>>>(proj_w, projT, DIM, DIM);
  cvt_transpose_kernel<<<(DIM * MLP_DIM + 255) / 256, 256, 0, stream>>>(fc1_w, fc1T, DIM, MLP_DIM);
  cvt_transpose_kernel<<<(MLP_DIM * DIM + 255) / 256, 256, 0, stream>>>(fc2_w, fc2T, MLP_DIM, DIM);

  // --- LN1 -> h (f16) ---
  ln_f16_kernel<<<TOKENS / 8, 256, 0, stream>>>(x, ln1_g, ln1_b, hln);

  // --- QKV GEMM: [16384,384] x [384,1152] -> f16 ---
  {
    dim3 grid(QKV_LD / 128, TOKENS / 128);
    gemm_wmma_kernel<QKV_LD, DIM, false, false, false>
        <<<grid, 128, 0, stream>>>(hln, qkvT, qkv_b, nullptr, nullptr, qkv_h);
  }

  // --- V transpose for attention ---
  transpose_v_kernel<<<(BATCH * HEADS * HEAD_DIM * SEQ) / 256, 256, 0, stream>>>(qkv_h, vtb);

  // --- flash attention ---
  {
    dim3 grid(SEQ / 64, BATCH * HEADS);
    flash_attn_kernel<<<grid, 128, 0, stream>>>(qkv_h, vtb, att_o);
  }

  // --- proj GEMM + residual: x1 = x + attn_out @ proj_w + b ---
  {
    dim3 grid(DIM / 128, TOKENS / 128);
    gemm_wmma_kernel<DIM, DIM, true, true, false>
        <<<grid, 128, 0, stream>>>(att_o, projT, proj_b, x, x1, nullptr);
  }

  // --- LN2 -> h (f16) ---
  ln_f16_kernel<<<TOKENS / 8, 256, 0, stream>>>(x1, ln2_g, ln2_b, hln);

  // --- fc1 GEMM + GELU -> f16 ---
  {
    dim3 grid(MLP_DIM / 128, TOKENS / 128);
    gemm_wmma_kernel<MLP_DIM, DIM, false, false, true>
        <<<grid, 128, 0, stream>>>(hln, fc1T, fc1_b, nullptr, nullptr, hmlp);
  }

  // --- fc2 GEMM + residual -> out fp32 ---
  {
    dim3 grid(DIM / 128, TOKENS / 128);
    gemm_wmma_kernel<DIM, MLP_DIM, true, true, false>
        <<<grid, 128, 0, stream>>>(hmlp, fc2T, fc2_b, x1, out, nullptr);
  }
}
